// SpatialContextEncoderEfficient_35210141893290
// MI455X (gfx1250) — matrix operand order repaired and checked by reading, not compiled, not run on hardware
//
#include <hip/hip_runtime.h>

// ---------------------------------------------------------------------------
// SpatialContextEncoder for MI455X (gfx1250, wave32, WMMA)
//   x : (8,1024,64,64) f32   W : (1024,1049) f32   b : (1024,) f32
//   out = relu( W @ concat[x, l2norm(localcorr25(x))] + b )  -> (8,1024,64,64)
// Dominant cost: 32768 x 1049 x 1024 GEMM (~70 GFLOP) -> v_wmma_f32_16x16x32_bf16
// GEMM: 8 waves/block, wave tile 32(M)x64(N) = 2x4 frags, 8 WMMA / 12 b128 loads
// Workspace layout: F (bf16, 32768 x 1056) | Wb (bf16, 1024 x 1056)  ~71.4 MB
// ---------------------------------------------------------------------------

typedef __attribute__((ext_vector_type(16))) __bf16    v16bf;
typedef __attribute__((ext_vector_type(8)))  float     v8f;
typedef __attribute__((ext_vector_type(8)))  unsigned  v8u;

#define KPAD   1056          // 1049 padded to multiple of 32
#define NPIX   4096          // 64*64
#define NCH    1024
#define NOUT   1024
#define NBATCH 8
#define MTOT   (NBATCH * NPIX)   // 32768

__device__ __forceinline__ unsigned short f2bf(float f) {
    union { float f; unsigned u; } v; v.f = f;
    unsigned r = v.u + 0x7FFFu + ((v.u >> 16) & 1u);   // round-to-nearest-even
    return (unsigned short)(r >> 16);
}

// ---------------------------------------------------------------------------
// Kernel 1: transpose x (b,c,pixel) -> F[m][c] bf16, via LDS 64x64 tile.
// grid = 8 * 16 * 64 = 8192 blocks, 256 threads
// ---------------------------------------------------------------------------
__global__ __launch_bounds__(256) void pack_x_kernel(const float* __restrict__ x,
                                                     unsigned short* __restrict__ F) {
    __shared__ float tile[64][65];
    int bid = blockIdx.x;
    int b   = bid >> 10;          // 16*64 tiles per batch
    int ct  = (bid >> 6) & 15;    // channel tile
    int pt  = bid & 63;           // pixel tile
    int c0 = ct * 64, p0 = pt * 64;
    const float* xb = x + (size_t)b * NCH * NPIX;
#pragma unroll
    for (int rep = 0; rep < 16; ++rep) {
        int idx = rep * 256 + threadIdx.x;
        int i = idx >> 6, j = idx & 63;                 // i: channel, j: pixel
        tile[i][j] = xb[(size_t)(c0 + i) * NPIX + p0 + j];
    }
    __syncthreads();
#pragma unroll
    for (int rep = 0; rep < 16; ++rep) {
        int idx = rep * 256 + threadIdx.x;
        int i = idx >> 6, j = idx & 63;                 // i: pixel, j: channel
        F[(size_t)(b * NPIX + p0 + i) * KPAD + c0 + j] = f2bf(tile[j][i]);
    }
}

// ---------------------------------------------------------------------------
// Kernel 2: 5x5 local self-correlation + l2norm -> F[m][1024..1055]
// one workgroup per (b, y) row; 8 waves split channels; LDS atomics reduce.
// grid = 8*64 = 512 blocks, 256 threads
// ---------------------------------------------------------------------------
__global__ __launch_bounds__(256) void corr_kernel(const float* __restrict__ x,
                                                   unsigned short* __restrict__ F) {
    __shared__ float acc[25][64];
    __shared__ float rows[8][5][68];    // per-wave padded row cache (x in [-2,65])
    int b = blockIdx.x >> 6;
    int y = blockIdx.x & 63;
    int tid = threadIdx.x;
    int w = tid >> 5, lane = tid & 31;

    for (int i = tid; i < 25 * 64; i += 256) (&acc[0][0])[i] = 0.f;
    __syncthreads();

    float ac[25][2];
#pragma unroll
    for (int j = 0; j < 25; ++j) { ac[j][0] = 0.f; ac[j][1] = 0.f; }

    for (int c = w; c < NCH; c += 8) {
        const float* xc = x + (size_t)(b * NCH + c) * NPIX;
        for (int idx = lane; idx < 5 * 68; idx += 32) {
            int r  = idx / 68;
            int xx = (idx % 68) - 2;
            int yy = y + r - 2;
            float v = 0.f;
            if ((unsigned)yy < 64u && (unsigned)xx < 64u) v = xc[yy * 64 + xx];
            rows[w][r][idx % 68] = v;
        }
        __syncthreads();
        float c0 = rows[w][2][2 + lane];
        float c1 = rows[w][2][2 + lane + 32];
#pragma unroll
        for (int dy = 0; dy < 5; ++dy)
#pragma unroll
            for (int dx = 0; dx < 5; ++dx) {
                int j = dy * 5 + dx;     // rows[][dy][x + dx] == x[y+dy-2][x+dx-2]
                ac[j][0] += rows[w][dy][lane + dx]      * c0;
                ac[j][1] += rows[w][dy][lane + 32 + dx] * c1;
            }
        __syncthreads();
    }
#pragma unroll
    for (int j = 0; j < 25; ++j) {
        atomicAdd(&acc[j][lane],      ac[j][0]);   // ds_add_f32
        atomicAdd(&acc[j][lane + 32], ac[j][1]);
    }
    __syncthreads();
    if (tid < 64) {
        int xp = tid;
        float s = 1e-6f;
        float v[25];
#pragma unroll
        for (int j = 0; j < 25; ++j) { v[j] = acc[j][xp]; s += v[j] * v[j]; }
        float inv = 1.0f / sqrtf(s);
        unsigned short* p = F + (size_t)(b * NPIX + y * 64 + xp) * KPAD + NCH;
#pragma unroll
        for (int j = 0; j < 25; ++j) p[j] = f2bf(v[j] * inv);
#pragma unroll
        for (int j = 25; j < 32; ++j) p[j] = 0;    // zero K-pad 1049..1055
    }
}

// ---------------------------------------------------------------------------
// Kernel 3: W (1024 x 1049 f32) -> Wb (1024 x 1056 bf16, zero-padded)
// ---------------------------------------------------------------------------
__global__ __launch_bounds__(256) void pack_w_kernel(const float* __restrict__ W,
                                                     unsigned short* __restrict__ Wb) {
    int idx = blockIdx.x * 256 + threadIdx.x;
    if (idx >= NOUT * KPAD) return;
    int o = idx / KPAD, k = idx % KPAD;
    Wb[idx] = (k < 1049) ? f2bf(W[(size_t)o * 1049 + k]) : (unsigned short)0;
}

// ---------------------------------------------------------------------------
// Kernel 4: GEMM  out[m][o] = relu( sum_k F[m][k] * Wb[o][k] + bias[o] )
// 8 waves / block arranged 4(M) x 2(N); block tile 128 x 128.
// Wave tile 32(M) x 64(N): 2 A-frags x 4 B-frags -> 8 WMMA per 12 b128 loads
//   (21.3 FLOP/byte from cache, ~64 acc VGPRs -> good occupancy).
// A 16x32 bf16 lane layout: lane<16 -> K {0..7,16..23}; lane>=16 -> K {8..15,24..31}
// B 32x16 bf16 lane layout: n = lane&15; lane<16 -> K {0..7,16..23}; lane>=16 -> K {8..15,24..31}
// D 16x16 f32: n = lane&15; rows (lane>=16? 8:0)+vgpr  (consecutive pixels)
// grid = (32768/128) * (1024/128) = 2048 blocks, 256 threads
// ---------------------------------------------------------------------------
__global__ __launch_bounds__(256) void gemm_kernel(const unsigned short* __restrict__ F,
                                                   const unsigned short* __restrict__ Wb,
                                                   const float* __restrict__ bias,
                                                   float* __restrict__ out) {
    int tile = blockIdx.x;
    int mt = tile >> 3, nt = tile & 7;
    int wid  = threadIdx.x >> 5;
    int lane = threadIdx.x & 31;
    int wm = wid & 3, wn = wid >> 2;
    int mBase = mt * 128 + wm * 32;
    int nBase = nt * 128 + wn * 64;
    int l15 = lane & 15;
    int hi  = lane >> 4;

    const unsigned short* arow[2];
    const unsigned short* brow[4];
#pragma unroll
    for (int i = 0; i < 2; ++i) arow[i] = F  + (size_t)(mBase + 16 * i + l15) * KPAD;
#pragma unroll
    for (int j = 0; j < 4; ++j) brow[j] = Wb + (size_t)(nBase + 16 * j + l15) * KPAD;

    v8f acc[2][4] = {};

    for (int kb = 0; kb < KPAD; kb += 32) {
        v16bf a[2], bm[4];
#pragma unroll
        for (int i = 0; i < 2; ++i) {
            const uint4* p0 = (const uint4*)(arow[i] + kb +      hi * 8);
            const uint4* p1 = (const uint4*)(arow[i] + kb + 16 + hi * 8);
            uint4 u0 = *p0, u1 = *p1;
            v8u ra = { u0.x, u0.y, u0.z, u0.w, u1.x, u1.y, u1.z, u1.w };
            a[i] = __builtin_bit_cast(v16bf, ra);
        }
#pragma unroll
        for (int j = 0; j < 4; ++j) {
            const uint4* q = (const uint4*)(brow[j] + kb + hi * 8);
            uint4 q0 = q[0], q1 = q[2];
            v8u rb = { q0.x, q0.y, q0.z, q0.w, q1.x, q1.y, q1.z, q1.w };
            bm[j] = __builtin_bit_cast(v16bf, rb);
        }
#pragma unroll
        for (int i = 0; i < 2; ++i)
#pragma unroll
            for (int j = 0; j < 4; ++j)
                acc[i][j] = __builtin_amdgcn_wmma_f32_16x16x32_bf16(
                    false, a[i], false, bm[j], (short)0, acc[i][j], false, false);
    }

#pragma unroll
    for (int i = 0; i < 2; ++i) {
        int rowBase = mBase + i * 16 + hi * 8;      // global m of first D row
        int bidx = rowBase >> 12;                   // batch
        int pix  = rowBase & (NPIX - 1);            // pixel (128 | 4096 => same batch)
#pragma unroll
        for (int j = 0; j < 4; ++j) {
            int o = nBase + j * 16 + l15;
            float bv = bias[o];
            float* op = out + (size_t)bidx * NOUT * NPIX + (size_t)o * NPIX + pix;
            float vals[8];
#pragma unroll
            for (int e = 0; e < 8; ++e) vals[e] = fmaxf(acc[i][j][e] + bv, 0.f);
            float4 lo4 = make_float4(vals[0], vals[1], vals[2], vals[3]);
            float4 hi4 = make_float4(vals[4], vals[5], vals[6], vals[7]);
            *(float4*)(op)     = lo4;
            *(float4*)(op + 4) = hi4;
        }
    }
}

// ---------------------------------------------------------------------------
extern "C" void kernel_launch(void* const* d_in, const int* in_sizes, int n_in,
                              void* d_out, int out_size, void* d_ws, size_t ws_size,
                              hipStream_t stream) {
    const float* x    = (const float*)d_in[0];
    const float* W    = (const float*)d_in[1];
    const float* bias = (const float*)d_in[2];
    float* out = (float*)d_out;

    unsigned short* F  = (unsigned short*)d_ws;              // 32768*1056*2 = 69.2 MB
    unsigned short* Wb = F + (size_t)MTOT * KPAD;            // 1024*1056*2  =  2.2 MB

    pack_x_kernel<<<NBATCH * 16 * 64, 256, 0, stream>>>(x, F);
    corr_kernel  <<<NBATCH * 64,      256, 0, stream>>>(x, F);
    pack_w_kernel<<<(NOUT * KPAD + 255) / 256, 256, 0, stream>>>(W, Wb);
    gemm_kernel  <<<(MTOT / 128) * (NOUT / 128), 256, 0, stream>>>(F, Wb, bias, out);
}
